// Net_20126216749340
// MI455X (gfx1250) — compile-verified, matrix operand order checked by reading, not run on hardware
//
#include <hip/hip_runtime.h>
#include <math.h>

// ---------------- problem constants (from reference) ----------------
#define BB   128      // batch (graphs)
#define NPG  256      // nodes per graph, stage 0
#define DEG  16       // edges per node
#define FF   128      // input features
#define HH   128      // hidden
#define NCLS 10       // classes
#define BM   1152     // block-merge dim = 9*H
#define MK   3        // stages
#define EE   ((long)BB * NPG * DEG)   // 524288 edges (constant across stages)

typedef __attribute__((ext_vector_type(16))) __bf16 v16bf;
typedef __attribute__((ext_vector_type(8)))  float  v8f;

// round-to-nearest-even float -> bf16 bit pattern
__device__ __forceinline__ unsigned short f2bfbits(float f) {
    unsigned int u = __builtin_bit_cast(unsigned int, f);
    return (unsigned short)((u + 0x7FFFu + ((u >> 16) & 1u)) >> 16);
}

// =====================================================================
// Operand packing: convert fp32 matrices into bf16 WMMA-fragment order.
// Fragment layout (ISA 7.12.2, 16-bit A 16x32): lane L (half = L>>4) holds
// two contiguous k-runs:  e in 0..7  -> k = half*8 + e
//                         e in 8..15 -> k = 16 + half*8 + (e-8)
// Packed buffer: slot index ((tTile*Kb + kb)*32 + lane), 16 bf16 = 32 bytes.
// =====================================================================

// Pack activations A[M,K] (row-major, leading dim lda). One thread per slot.
__global__ void pack_a_kernel(const float* __restrict__ A, int lda, int Kb,
                              unsigned int* __restrict__ dstp, long total) {
    long t = (long)blockIdx.x * blockDim.x + threadIdx.x;
    if (t >= total) return;
    int  lane = (int)(t & 31);
    long rest = t >> 5;
    int  kb   = (int)(rest % Kb);
    long tM   = rest / Kb;
    int  half = (lane >> 4) & 1;
    int  mrow = (int)(tM * 16 + (lane & 15));
    int  k0   = kb * 32;
    const float* a0 = A + (long)mrow * lda + k0 + half * 8;  // 8 contiguous floats
    const float* a1 = a0 + 16;                               // 8 contiguous floats
    unsigned int wds[8];
#pragma unroll
    for (int j = 0; j < 4; ++j) {
        wds[j]     = (unsigned)f2bfbits(a0[2*j]) | ((unsigned)f2bfbits(a0[2*j+1]) << 16);
        wds[4 + j] = (unsigned)f2bfbits(a1[2*j]) | ((unsigned)f2bfbits(a1[2*j+1]) << 16);
    }
    unsigned int* o = dstp + t * 8;
    ((uint4*)o)[0] = make_uint4(wds[0], wds[1], wds[2], wds[3]);
    ((uint4*)o)[1] = make_uint4(wds[4], wds[5], wds[6], wds[7]);
}

// Pack weights W[K,Nvalid] (row-major, leading dim ldw) into B-fragment order,
// zero-filling padded columns. One thread per slot; strided gather happens here
// exactly once per GEMM instead of once per output tile.
__global__ void pack_b_kernel(const float* __restrict__ W, int ldw, int Kb, int Nvalid,
                              unsigned int* __restrict__ dstp, long total) {
    long t = (long)blockIdx.x * blockDim.x + threadIdx.x;
    if (t >= total) return;
    int  lane = (int)(t & 31);
    long rest = t >> 5;
    int  kb   = (int)(rest % Kb);
    long tN   = rest / Kb;
    int  half = (lane >> 4) & 1;
    int  ncol = (int)(tN * 16 + (lane & 15));
    int  k0   = kb * 32;
    unsigned int wds[8];
    if (ncol < Nvalid) {
        const float* w0 = W + (long)(k0 + half * 8) * ldw + ncol;
#pragma unroll
        for (int j = 0; j < 4; ++j) {
            float f0 = w0[(long)(2*j)     * ldw];
            float f1 = w0[(long)(2*j + 1) * ldw];
            float g0 = w0[(long)(16 + 2*j)     * ldw];
            float g1 = w0[(long)(16 + 2*j + 1) * ldw];
            wds[j]     = (unsigned)f2bfbits(f0) | ((unsigned)f2bfbits(f1) << 16);
            wds[4 + j] = (unsigned)f2bfbits(g0) | ((unsigned)f2bfbits(g1) << 16);
        }
    } else {
#pragma unroll
        for (int j = 0; j < 8; ++j) wds[j] = 0u;
    }
    unsigned int* o = dstp + t * 8;
    ((uint4*)o)[0] = make_uint4(wds[0], wds[1], wds[2], wds[3]);
    ((uint4*)o)[1] = make_uint4(wds[4], wds[5], wds[6], wds[7]);
}

// =====================================================================
// WMMA GEMM on packed operands. One wave computes a 16x32 slab (two 16x16
// N-tiles sharing the A fragment -> 2 wmma per k-step, no branches in loop).
// =====================================================================
__device__ __forceinline__ void store_tile(float* __restrict__ C, int ldc, const v8f& acc,
                                           int mbase, int half, int ncol, int Nvalid,
                                           const float* __restrict__ bias, int act) {
    if (ncol >= Nvalid) return;
    float bv = bias ? bias[ncol] : 0.f;
#pragma unroll
    for (int r = 0; r < 8; ++r) {       // D layout: VGPR r -> M = r + 8*half
        float v = acc[r] + bv;
        if (act) v = fmaxf(v, 0.f);
        C[(long)(mbase + half * 8 + r) * ldc + ncol] = v;
    }
}

__global__ void gemm_wmma_pk_kernel(const unsigned int* __restrict__ Apk,
                                    const unsigned int* __restrict__ Bpk,
                                    const float* __restrict__ bias,
                                    float* __restrict__ C, int ldc,
                                    int M, int Kb, int Nvalid, int act,
                                    int tilesN, int tilesNP) {
    int gid  = blockIdx.x * blockDim.x + threadIdx.x;
    int wave = gid >> 5;
    int lane = threadIdx.x & 31;
    int tilesM = M >> 4;
    if (wave >= tilesM * tilesNP) return;
    int tM  = wave / tilesNP;
    int tNP = wave % tilesNP;
    int tN0 = tNP * 2;
    int tN1 = tN0 + 1;
    int tN1c = (tN1 < tilesN) ? tN1 : (tilesN - 1);   // clamp, not branch

    const v16bf* ap  = (const v16bf*)Apk + (long)tM  * Kb * 32 + lane;
    const v16bf* bp0 = (const v16bf*)Bpk + (long)tN0 * Kb * 32 + lane;
    const v16bf* bp1 = (const v16bf*)Bpk + (long)tN1c * Kb * 32 + lane;

    v8f acc0 = {0.f, 0.f, 0.f, 0.f, 0.f, 0.f, 0.f, 0.f};
    v8f acc1 = {0.f, 0.f, 0.f, 0.f, 0.f, 0.f, 0.f, 0.f};

#pragma unroll 2
    for (int kb = 0; kb < Kb; ++kb) {
        v16bf a  = ap [(long)kb * 32];
        v16bf b0 = bp0[(long)kb * 32];
        v16bf b1 = bp1[(long)kb * 32];
        acc0 = __builtin_amdgcn_wmma_f32_16x16x32_bf16(false, a, false, b0,
                                                       (short)0, acc0, false, false);
        acc1 = __builtin_amdgcn_wmma_f32_16x16x32_bf16(false, a, false, b1,
                                                       (short)0, acc1, false, false);
    }

    int half  = (lane >> 4) & 1;
    int mbase = tM * 16;
    int nlan  = lane & 15;
    store_tile(C, ldc, acc0, mbase, half, tN0 * 16 + nlan, Nvalid, bias, act);
    if (tN1 < tilesN)
        store_tile(C, ldc, acc1, mbase, half, tN1 * 16 + nlan, Nvalid, bias, act);
}

// ---------------- small utility kernels ----------------
__global__ void init_edges_kernel(const int* __restrict__ ei, int* __restrict__ s,
                                  int* __restrict__ d, float* __restrict__ w, long E) {
    long t = (long)blockIdx.x * blockDim.x + threadIdx.x;
    if (t >= E) return;
    s[t] = ei[t];
    d[t] = ei[E + t];
    w[t] = 1.0f;
}

__global__ void fill_f_kernel(float* __restrict__ a, float v, long n) {
    long t = (long)blockIdx.x * blockDim.x + threadIdx.x;
    if (t < n) a[t] = v;
}

__global__ void zero_i_kernel(int* __restrict__ a, long n) {
    long t = (long)blockIdx.x * blockDim.x + threadIdx.x;
    if (t < n) a[t] = 0;
}

__global__ void deg_acc_kernel(const int* __restrict__ dst, const float* __restrict__ w,
                               float* __restrict__ deg, long E) {
    long t = (long)blockIdx.x * blockDim.x + threadIdx.x;
    if (t >= E) return;
    atomicAdd(&deg[dst[t]], w[t]);
}

__global__ void dis_selfc_kernel(const float* __restrict__ deg, float* __restrict__ dis,
                                 float* __restrict__ selfc, long n) {
    long t = (long)blockIdx.x * blockDim.x + threadIdx.x;
    if (t >= n) return;
    float d = rsqrtf(deg[t]);
    dis[t] = d;
    selfc[t] = d * d;
}

__global__ void edge_norm_kernel(const int* __restrict__ src, const int* __restrict__ dst,
                                 const float* __restrict__ w, const float* __restrict__ dis,
                                 float* __restrict__ norm, long E) {
    long t = (long)blockIdx.x * blockDim.x + threadIdx.x;
    if (t >= E) return;
    norm[t] = w[t] * dis[src[t]] * dis[dst[t]];
}

// agg[dst] += norm * h[src], 4 features per thread (H % 4 == 0)
__global__ void edge_agg4_kernel(const int* __restrict__ src, const int* __restrict__ dst,
                                 const float* __restrict__ norm, const float* __restrict__ h,
                                 float* __restrict__ agg, long E, int H4, int H) {
    long t = (long)blockIdx.x * blockDim.x + threadIdx.x;
    long total = E * (long)H4;
    if (t >= total) return;
    long e  = t / H4;
    int  f4 = (int)(t % H4) * 4;
    float nw = norm[e];
    const float4 hv = *(const float4*)(h + (long)src[e] * H + f4);
    float* ap = agg + (long)dst[e] * H + f4;
    atomicAdd(ap + 0, nw * hv.x);
    atomicAdd(ap + 1, nw * hv.y);
    atomicAdd(ap + 2, nw * hv.z);
    atomicAdd(ap + 3, nw * hv.w);
}

// scalar variant (H == 1, pool scorer)
__global__ void edge_agg1_kernel(const int* __restrict__ src, const int* __restrict__ dst,
                                 const float* __restrict__ norm, const float* __restrict__ h,
                                 float* __restrict__ agg, long E) {
    long t = (long)blockIdx.x * blockDim.x + threadIdx.x;
    if (t >= E) return;
    atomicAdd(&agg[dst[t]], norm[t] * h[src[t]]);
}

// out[i, f] = act(agg[i,f] + selfc[i]*h[i,f] + bias[f]), leading dim ldc
__global__ void gcn_finish_kernel(const float* __restrict__ agg, const float* __restrict__ h,
                                  const float* __restrict__ selfc, const float* __restrict__ bias,
                                  float* __restrict__ out, int ldc, long n, int H, int act) {
    long t = (long)blockIdx.x * blockDim.x + threadIdx.x;
    long total = n * (long)H;
    if (t >= total) return;
    long i = t / H;
    int  f = (int)(t % H);
    float v = agg[t] + selfc[i] * h[t] + bias[f];
    if (act) v = fmaxf(v, 0.f);
    out[i * (long)ldc + f] = v;
}

// per-graph top-k by score; one block (256 threads) per graph
__global__ void topk_kernel(const float* __restrict__ score, int n_per, int k,
                            int* __restrict__ perm) {
    __shared__ float vals[256];
    __shared__ float sv[256];
    __shared__ int   si[256];
    int b = blockIdx.x, t = threadIdx.x;
    vals[t] = (t < n_per) ? score[(long)b * n_per + t] : -3.4e38f;
    __syncthreads();
    for (int it = 0; it < k; ++it) {
        sv[t] = vals[t];
        si[t] = t;
        __syncthreads();
        for (int s = 128; s > 0; s >>= 1) {
            if (t < s) {
                float v2 = sv[t + s];
                int   i2 = si[t + s];
                if (v2 > sv[t] || (v2 == sv[t] && i2 < si[t])) { sv[t] = v2; si[t] = i2; }
            }
            __syncthreads();
        }
        if (t == 0) {
            perm[(long)b * k + it] = b * n_per + si[0];
            vals[si[0]] = -3.4e38f;
        }
        __syncthreads();
    }
}

__global__ void gather_newx_kernel(const float* __restrict__ xt, const float* __restrict__ score,
                                   const int* __restrict__ perm, float* __restrict__ newx,
                                   long nk) {
    long t = (long)blockIdx.x * blockDim.x + threadIdx.x;
    long total = nk * (long)(BM / 4);
    if (t >= total) return;
    long i  = t / (BM / 4);
    int  f4 = (int)(t % (BM / 4)) * 4;
    int  p  = perm[i];
    float g = tanhf(score[p]);
    float4 v = *(const float4*)(xt + (long)p * BM + f4);
    float4 o; o.x = v.x * g; o.y = v.y * g; o.z = v.z * g; o.w = v.w * g;
    *(float4*)(newx + i * (long)BM + f4) = o;
}

__global__ void mark_kept_kernel(const int* __restrict__ perm, int* __restrict__ kept,
                                 int* __restrict__ nidx, long nk) {
    long t = (long)blockIdx.x * blockDim.x + threadIdx.x;
    if (t >= nk) return;
    int p = perm[t];
    kept[p] = 1;
    nidx[p] = (int)t;
}

__global__ void filter_edges_kernel(int* __restrict__ src, int* __restrict__ dst,
                                    float* __restrict__ w, const int* __restrict__ kept,
                                    const int* __restrict__ nidx, long E) {
    long t = (long)blockIdx.x * blockDim.x + threadIdx.x;
    if (t >= E) return;
    int s = src[t], d = dst[t];
    bool valid = (kept[s] != 0) && (kept[d] != 0);
    src[t] = valid ? nidx[s] : 0;
    dst[t] = valid ? nidx[d] : 0;
    w[t]   = valid ? w[t] : 0.f;
}

// global max pool over k nodes per graph -> MLP staging buffer AND d_out slice
__global__ void maxpool_kernel(const float* __restrict__ newx, int k,
                               float* __restrict__ result, int resLd, int colOff,
                               float* __restrict__ outp) {
    long t = (long)blockIdx.x * blockDim.x + threadIdx.x;
    long total = (long)BB * BM;
    if (t >= total) return;
    int b = (int)(t / BM);
    int f = (int)(t % BM);
    float m = -3.4e38f;
    for (int j = 0; j < k; ++j)
        m = fmaxf(m, newx[((long)b * k + j) * BM + f]);
    result[(long)b * resLd + colOff + f] = m;
    outp[t] = m;
}

__global__ void logsoftmax_kernel(const float* __restrict__ z, int ldz,
                                  float* __restrict__ out, int M, int N) {
    int b = blockIdx.x * blockDim.x + threadIdx.x;
    if (b >= M) return;
    float mx = -3.4e38f;
    for (int j = 0; j < N; ++j) mx = fmaxf(mx, z[(long)b * ldz + j]);
    float s = 0.f;
    for (int j = 0; j < N; ++j) s += expf(z[(long)b * ldz + j] - mx);
    float lse = logf(s);
    for (int j = 0; j < N; ++j) out[(long)b * N + j] = z[(long)b * ldz + j] - mx - lse;
}

// ---------------- host-side helpers ----------------
static inline unsigned gdim(long total, int block) {
    return (unsigned)((total + block - 1) / block);
}

// C[M,Nvalid] = act(A[M,K] @ W[K,Nvalid] + bias); packs both operands to bf16 first
static void run_gemm(hipStream_t st, const float* A, int lda, const float* W, int ldw,
                     const float* bias, float* C, int ldc, int M, int K, int Nvalid, int act,
                     unsigned int* Apk, unsigned int* Bpk) {
    int Kb      = K / 32;
    int tilesM  = M / 16;
    int tilesN  = (Nvalid + 15) / 16;
    int tilesNP = (tilesN + 1) / 2;
    long aslots = (long)tilesM * Kb * 32;
    long bslots = (long)tilesN * Kb * 32;
    pack_a_kernel<<<gdim(aslots, 256), 256, 0, st>>>(A, lda, Kb, Apk, aslots);
    pack_b_kernel<<<gdim(bslots, 256), 256, 0, st>>>(W, ldw, Kb, Nvalid, Bpk, bslots);
    long waves = (long)tilesM * tilesNP;
    gemm_wmma_pk_kernel<<<gdim(waves * 32, 256), 256, 0, st>>>(
        Apk, Bpk, bias, C, ldc, M, Kb, Nvalid, act, tilesN, tilesNP);
}

// one full GCN layer:  out = act(agg(norm * (A@W)) + selfc*(A@W) + bias)
static void run_gcn(hipStream_t st, const float* A, int lda, int K,
                    const float* W, const float* bias,
                    const int* src, const int* dst, const float* norm,
                    const float* selfc, long n, long E, int Hout,
                    float* htmp, float* agg, float* out, int ldc, int act,
                    unsigned int* Apk, unsigned int* Bpk) {
    run_gemm(st, A, lda, W, Hout, nullptr, htmp, Hout, (int)n, K, Hout, 0, Apk, Bpk);
    long tot = n * (long)Hout;
    fill_f_kernel<<<gdim(tot, 256), 256, 0, st>>>(agg, 0.f, tot);
    if (Hout % 4 == 0) {
        edge_agg4_kernel<<<gdim(E * (Hout / 4), 256), 256, 0, st>>>(
            src, dst, norm, htmp, agg, E, Hout / 4, Hout);
    } else {
        edge_agg1_kernel<<<gdim(E, 256), 256, 0, st>>>(src, dst, norm, htmp, agg, E);
    }
    gcn_finish_kernel<<<gdim(tot, 256), 256, 0, st>>>(agg, htmp, selfc, bias, out, ldc, n, Hout, act);
}

extern "C" void kernel_launch(void* const* d_in, const int* in_sizes, int n_in,
                              void* d_out, int out_size, void* d_ws, size_t ws_size,
                              hipStream_t stream) {
    (void)in_sizes; (void)n_in; (void)out_size; (void)ws_size;

    const float* x          = (const float*)d_in[0];
    const int*   edge_index = (const int*)d_in[1];
    /* d_in[2] = batch (unused: layout is fixed B x n_per) */
    const float* b1_W  = (const float*)d_in[3];
    const float* b1_b  = (const float*)d_in[4];
    const float* b2_W  = (const float*)d_in[5];
    const float* b2_b  = (const float*)d_in[6];
    const float* b3_W  = (const float*)d_in[7];
    const float* b3_b  = (const float*)d_in[8];
    const float* b21_W = (const float*)d_in[9];
    const float* b21_b = (const float*)d_in[10];
    const float* b22_W = (const float*)d_in[11];
    const float* b22_b = (const float*)d_in[12];
    const float* b23_W = (const float*)d_in[13];
    const float* b23_b = (const float*)d_in[14];
    const float* pool_W = (const float*)d_in[15];
    const float* pool_b = (const float*)d_in[16];
    const float* k_weight = (const float*)d_in[17];
    const float* lin1_W = (const float*)d_in[18];
    const float* lin1_b = (const float*)d_in[19];
    const float* lin2_W = (const float*)d_in[20];
    const float* lin2_b = (const float*)d_in[21];
    const float* lin3_W = (const float*)d_in[22];
    const float* lin3_b = (const float*)d_in[23];
    float* out = (float*)d_out;

    // ----- bump allocator over d_ws -----
    char* wsp = (char*)d_ws;
    size_t off = 0;
    auto alloc = [&](size_t bytes) -> void* {
        void* p = wsp + off;
        off = (off + bytes + 255) & ~(size_t)255;
        return p;
    };
    const long N0 = (long)BB * NPG;                 // 32768
    float* xt     = (float*)alloc((size_t)N0 * BM * 4);
    float* htmp   = (float*)alloc((size_t)N0 * HH * 4);
    float* agg    = (float*)alloc((size_t)N0 * HH * 4);
    float* score  = (float*)alloc((size_t)N0 * 4);
    float* deg    = (float*)alloc((size_t)N0 * 4);
    float* dis    = (float*)alloc((size_t)N0 * 4);
    float* selfc  = (float*)alloc((size_t)N0 * 4);
    float* norm   = (float*)alloc((size_t)EE * 4);
    float* w      = (float*)alloc((size_t)EE * 4);
    int*   cur_src= (int*)alloc((size_t)EE * 4);
    int*   cur_dst= (int*)alloc((size_t)EE * 4);
    float* newx   = (float*)alloc((size_t)BB * 128 * BM * 4);
    int*   perm   = (int*)alloc((size_t)BB * 128 * 4);
    int*   kept   = (int*)alloc((size_t)N0 * 4);
    int*   nidx   = (int*)alloc((size_t)N0 * 4);
    float* result = (float*)alloc((size_t)BB * (MK * BM) * 4);
    float* z1     = (float*)alloc((size_t)BB * BM * 4);
    float* z2     = (float*)alloc((size_t)BB * HH * 4);
    float* z3     = (float*)alloc((size_t)BB * 64 * 4);
    float* z4     = (float*)alloc((size_t)BB * 16 * 4);
    // packed bf16 operand staging (max: A = 32768 x 1152, B = 3456 x 1152)
    unsigned int* Apk = (unsigned int*)alloc((size_t)N0 * BM * 2);
    unsigned int* Bpk = (unsigned int*)alloc((size_t)(MK * BM) * BM * 2);

    // ----- mutable edge copy, w = 1 -----
    init_edges_kernel<<<gdim(EE, 256), 256, 0, stream>>>(edge_index, cur_src, cur_dst, w, EE);

    const int n_pers[3] = {NPG, 128, 64};
    const int ks[3]     = {128, 64, 32};
    const int b2cnt[3]  = {0, 1, 2};
    const int b2off[3][2] = {{0, 0}, {0, 0}, {1, 2}};

    const float* cur_x = x;
    int cur_lda = FF;

    for (int stage = 0; stage < 3; ++stage) {
        long n_per = n_pers[stage];
        long n = (long)BB * n_per;

        // ----- _geom -----
        fill_f_kernel<<<gdim(n, 256), 256, 0, stream>>>(deg, 1.0f, n);
        deg_acc_kernel<<<gdim(EE, 256), 256, 0, stream>>>(cur_dst, w, deg, EE);
        dis_selfc_kernel<<<gdim(n, 256), 256, 0, stream>>>(deg, dis, selfc, n);
        edge_norm_kernel<<<gdim(EE, 256), 256, 0, stream>>>(cur_src, cur_dst, w, dis, norm, EE);

        // ----- _propagate: 3 aspects -> concat into xt[:, 0..BM) -----
        const float *W1, *B1, *W2, *B2p, *W3, *B3;
        int K_in;
        if (stage == 0) { W1=b1_W;  B1=b1_b;  W2=b2_W;  B2p=b2_b;  W3=b3_W;  B3=b3_b;  K_in=FF; }
        else            { W1=b21_W; B1=b21_b; W2=b22_W; B2p=b22_b; W3=b23_W; B3=b23_b; K_in=BM; }
        long W1stride = (long)K_in * HH;

        int col = 0;
        for (int i = 0; i < 3; ++i) {
            run_gcn(stream, cur_x, cur_lda, K_in, W1 + i * W1stride, B1 + i * HH,
                    cur_src, cur_dst, norm, selfc, n, EE, HH, htmp, agg, xt + col, BM, 1,
                    Apk, Bpk);
            int curcol = col; col += HH;
            for (int jj = 0; jj < b2cnt[i]; ++jj) {
                int j = b2off[i][jj];
                run_gcn(stream, xt + curcol, BM, HH, W2 + (long)j * HH * HH, B2p + j * HH,
                        cur_src, cur_dst, norm, selfc, n, EE, HH, htmp, agg, xt + col, BM, 1,
                        Apk, Bpk);
                curcol = col; col += HH;
            }
            run_gcn(stream, xt + curcol, BM, HH, W3 + (long)i * HH * HH, B3 + i * HH,
                    cur_src, cur_dst, norm, selfc, n, EE, HH, htmp, agg, xt + col, BM, 1,
                    Apk, Bpk);
            col += HH;
        }

        // ----- _sagpool -----
        run_gcn(stream, xt, BM, BM, pool_W, pool_b,
                cur_src, cur_dst, norm, selfc, n, EE, 1, htmp, agg, score, 1, 0, Apk, Bpk);
        int k = ks[stage];
        long nk = (long)BB * k;
        topk_kernel<<<BB, 256, 0, stream>>>(score, (int)n_per, k, perm);
        gather_newx_kernel<<<gdim(nk * (BM / 4), 256), 256, 0, stream>>>(xt, score, perm, newx, nk);
        zero_i_kernel<<<gdim(n, 256), 256, 0, stream>>>(kept, n);
        zero_i_kernel<<<gdim(n, 256), 256, 0, stream>>>(nidx, n);
        mark_kept_kernel<<<gdim(nk, 256), 256, 0, stream>>>(perm, kept, nidx, nk);
        filter_edges_kernel<<<gdim(EE, 256), 256, 0, stream>>>(cur_src, cur_dst, w, kept, nidx, EE);

        // ----- global max pool -----
        float* out_slice = out + 1280 + (long)stage * BB * BM;
        maxpool_kernel<<<gdim((long)BB * BM, 256), 256, 0, stream>>>(
            newx, k, result, MK * BM, stage * BM, out_slice);

        cur_x = newx;
        cur_lda = BM;
    }

    // ----- MLP head -----
    run_gemm(stream, result, MK * BM, k_weight, BM, nullptr, z1, BM, BB, MK * BM, BM, 0, Apk, Bpk);
    run_gemm(stream, z1, BM, lin1_W, HH, lin1_b, z2, HH, BB, BM, HH, 1, Apk, Bpk);
    run_gemm(stream, z2, HH, lin2_W, 64, lin2_b, z3, 64, BB, HH, 64, 1, Apk, Bpk);
    run_gemm(stream, z3, 64, lin3_W, NCLS, lin3_b, z4, 16, BB, 64, NCLS, 0, Apk, Bpk);
    logsoftmax_kernel<<<gdim(BB, 128), 128, 0, stream>>>(z4, 16, out, BB, NCLS);
}
